// TemporalAttention_89163521065225
// MI455X (gfx1250) — matrix operand order repaired and checked by reading, not compiled, not run on hardware
//
#include <hip/hip_runtime.h>

typedef __bf16 bf16;
typedef __attribute__((ext_vector_type(16))) __bf16 v16bf;
typedef __attribute__((ext_vector_type(8)))  __bf16 v8bf;
typedef __attribute__((ext_vector_type(8)))  float  v8f;

#define NDIM 768
#define NH   12
#define NKV  4
#define HD   64
#define TDIM 128
#define SDIM 256
#define NROWS (TDIM * SDIM)      // 32768 rows in (t, s) order
#define KSTEPS (NDIM / 32)       // 24

static __device__ __forceinline__ v8f vzero8() {
  v8f v;
#pragma unroll
  for (int i = 0; i < 8; ++i) v[i] = 0.0f;
  return v;
}

static __device__ __forceinline__ v8f wmma_bf16(v16bf a, v16bf b, v8f c) {
  return __builtin_amdgcn_wmma_f32_16x16x32_bf16(false, a, false, b, (short)0, c,
                                                 false, false);
}

// A-operand tile (16 rows x 32 k) from row-major bf16; stride in elems, %8==0.
static __device__ __forceinline__ v16bf load_a_bf16(const bf16* base, int stride) {
  int lane = threadIdx.x & 31;
  const bf16* p = base + (size_t)(lane & 15) * stride + ((lane >> 4) << 3);
  v8bf lo = *(const v8bf*)p;
  v8bf hi = *(const v8bf*)(p + 16);
  v16bf r;
#pragma unroll
  for (int i = 0; i < 8; ++i) { r[i] = lo[i]; r[8 + i] = hi[i]; }
  return r;
}

// B-operand tile (32 k x 16 n) where the source is row-major [n][k] (transposed
// operand). Lane l: n = l&15; k = (l>>4)*8 .. +7 and 16+(l>>4)*8 .. +7.
static __device__ __forceinline__ v16bf load_b_bf16(const bf16* base, int stride) {
  int lane = threadIdx.x & 31;
  const bf16* p = base + (size_t)(lane & 15) * stride + ((lane >> 4) << 3);
  v8bf lo = *(const v8bf*)p;
  v8bf hi = *(const v8bf*)(p + 16);
  v16bf r;
#pragma unroll
  for (int i = 0; i < 8; ++i) { r[i] = lo[i]; r[8 + i] = hi[i]; }
  return r;
}

// A-operand tile from row-major fp32, converting to bf16 on the fly.
static __device__ __forceinline__ v16bf load_a_f32(const float* base, int stride) {
  int lane = threadIdx.x & 31;
  const float* p = base + (size_t)(lane & 15) * stride + ((lane >> 4) << 3);
  float4 a0 = *(const float4*)(p + 0);
  float4 a1 = *(const float4*)(p + 4);
  float4 b0 = *(const float4*)(p + 16);
  float4 b1 = *(const float4*)(p + 20);
  float t[16] = {a0.x, a0.y, a0.z, a0.w, a1.x, a1.y, a1.z, a1.w,
                 b0.x, b0.y, b0.z, b0.w, b1.x, b1.y, b1.z, b1.w};
  v16bf r;
#pragma unroll
  for (int i = 0; i < 16; ++i) r[i] = (bf16)t[i];
  return r;
}

// ---------------------------------------------------------------------------
// Kernel 0: fp32 [K][N] -> bf16 transposed [N][K]
// ---------------------------------------------------------------------------
__global__ void k_tcvt(const float* __restrict__ src, bf16* __restrict__ dst,
                       int K, int N) {
  int idx = blockIdx.x * 256 + threadIdx.x;
  if (idx >= K * N) return;
  int k = idx / N, n = idx - k * N;
  dst[(size_t)n * K + k] = (bf16)src[idx];
}

// ---------------------------------------------------------------------------
// Kernel 1: QKV projection + fused per-head RMSNorm.
// One wave computes a 64-row x 64-col (one head) tile: each B tile feeds 4
// independent WMMAs (4 row-tiles) -> ~2x fewer loads per WMMA and 4 parallel
// accumulation chains per s_wait.
//   Q  -> [s][h][t][d]   (bf16)
//   K  -> [s][hk][t][d]  (bf16)
//   V  -> [s][hk][d][t]  (bf16, transposed so PV B-operand is row-major)
// ---------------------------------------------------------------------------
__global__ void __launch_bounds__(32) k_qkv(
    const float* __restrict__ x,
    const bf16* __restrict__ WqT, const bf16* __restrict__ WkT,
    const bf16* __restrict__ WvT,
    const float* __restrict__ qw, const float* __restrict__ kw,
    bf16* __restrict__ Q, bf16* __restrict__ K, bf16* __restrict__ Vt) {
  int r0 = blockIdx.x << 6;       // rows r0..r0+63; row = t*SDIM + s (same t)
  int hh = blockIdx.y;            // 0..11 Q heads, 12..15 K heads, 16..19 V heads
  int t  = r0 >> 8;
  int s0 = r0 & (SDIM - 1);
  const bf16* WT; int kind, hloc;
  if (hh < NH)            { WT = WqT; kind = 0; hloc = hh; }
  else if (hh < NH + NKV) { WT = WkT; kind = 1; hloc = hh - NH; }
  else                    { WT = WvT; kind = 2; hloc = hh - NH - NKV; }
  int colbase = hloc * HD;

  v8f acc[4][4];
#pragma unroll
  for (int rt = 0; rt < 4; ++rt)
#pragma unroll
    for (int nt = 0; nt < 4; ++nt) acc[rt][nt] = vzero8();

  int lane = threadIdx.x & 31;
  for (int kk = 0; kk < KSTEPS; ++kk) {
    if (kk + 1 < KSTEPS)   // hint next k-slice of A (gfx1250 global_prefetch_b8)
      __builtin_prefetch(x + (size_t)(r0 + lane) * NDIM + (kk + 1) * 32, 0, 3);
    v16bf a[4];
#pragma unroll
    for (int rt = 0; rt < 4; ++rt)
      a[rt] = load_a_f32(x + (size_t)(r0 + 16 * rt) * NDIM + kk * 32, NDIM);
#pragma unroll
    for (int nt = 0; nt < 4; ++nt) {
      v16bf b = load_b_bf16(WT + (size_t)(colbase + nt * 16) * NDIM + kk * 32, NDIM);
#pragma unroll
      for (int rt = 0; rt < 4; ++rt)
        acc[rt][nt] = wmma_bf16(a[rt], b, acc[rt][nt]);
    }
  }

  int half = lane >> 4, col = lane & 15;

  if (kind < 2) {  // RMSNorm over the 64 columns (= head_dim) of each row
    float wv[4];
    const float* w = (kind == 0) ? qw : kw;
#pragma unroll
    for (int nt = 0; nt < 4; ++nt) wv[nt] = w[nt * 16 + col];
#pragma unroll
    for (int rt = 0; rt < 4; ++rt)
#pragma unroll
      for (int j = 0; j < 8; ++j) {
        float ss = 0.0f;
#pragma unroll
        for (int nt = 0; nt < 4; ++nt) { float v = acc[rt][nt][j]; ss += v * v; }
        ss += __shfl_xor(ss, 1, 32);
        ss += __shfl_xor(ss, 2, 32);
        ss += __shfl_xor(ss, 4, 32);
        ss += __shfl_xor(ss, 8, 32);  // reduce within each 16-lane half (one row)
        float rms = rsqrtf(ss * (1.0f / HD) + 1e-6f);
#pragma unroll
        for (int nt = 0; nt < 4; ++nt) acc[rt][nt][j] *= rms * wv[nt];
      }
  }

#pragma unroll
  for (int rt = 0; rt < 4; ++rt)
#pragma unroll
    for (int j = 0; j < 8; ++j) {
      int m = rt * 16 + j + (half << 3);
      int s = s0 + m;
#pragma unroll
      for (int nt = 0; nt < 4; ++nt) {
        int c = nt * 16 + col;
        bf16 val = (bf16)acc[rt][nt][j];
        if (kind == 0)
          Q[(((size_t)s * NH + hloc) * TDIM + t) * HD + c] = val;
        else if (kind == 1)
          K[(((size_t)s * NKV + hloc) * TDIM + t) * HD + c] = val;
        else
          Vt[(((size_t)s * NKV + hloc) * HD + c) * TDIM + t] = val;
      }
    }
}

// ---------------------------------------------------------------------------
// Kernel 2: causal attention with softcap, one workgroup per (s, head).
// 4 waves x 32 query rows. K and V^T are staged once into LDS (32 KB) so all
// B-operand gathers hit LDS instead of re-reading global 8x. 96 KB dynamic LDS.
// ---------------------------------------------------------------------------
__global__ void __launch_bounds__(128) k_attn(
    const bf16* __restrict__ Q, const bf16* __restrict__ K,
    const bf16* __restrict__ Vt, bf16* __restrict__ AO) {
  extern __shared__ char smembase[];
  float* smem = (float*)smembase;
  int s = blockIdx.x, h = blockIdx.y;
  int hk = h / (NH / NKV);           // GQA: query head -> kv head
  int wave = threadIdx.x >> 5, lane = threadIdx.x & 31;
  int half = lane >> 4, col = lane & 15;
  const bf16* Qh = Q  + ((size_t)s * NH  + h)  * TDIM * HD;
  const bf16* Kh = K  + ((size_t)s * NKV + hk) * TDIM * HD;
  const bf16* Vh = Vt + ((size_t)s * NKV + hk) * HD * TDIM;
  float* sc = smem + wave * (32 * TDIM);            // 64 KB: per-wave 32x128
  bf16* Ks = (bf16*)(smembase + 4 * 32 * TDIM * sizeof(float));  // 128x64 bf16
  bf16* Vs = Ks + TDIM * HD;                        // 64x128 bf16

  {  // cooperative staging: 2 x 16 KB, 8 float4 per thread per array
    const float4* srcK = (const float4*)Kh;
    const float4* srcV = (const float4*)Vh;
    float4* dstK = (float4*)Ks;
    float4* dstV = (float4*)Vs;
#pragma unroll
    for (int i = 0; i < (TDIM * HD * 2 / 16) / 128; ++i) {
      int idx = i * 128 + threadIdx.x;
      dstK[idx] = srcK[idx];
      dstV[idx] = srcV[idx];
    }
  }

  v16bf aq[2][2];
#pragma unroll
  for (int i = 0; i < 2; ++i)
#pragma unroll
    for (int kk = 0; kk < 2; ++kk)
      aq[i][kk] = load_a_bf16(Qh + (wave * 32 + 16 * i) * HD + kk * 32, HD);

  __syncthreads();

  // ---- scores = softcap(scale * Q K^T) with causal mask -------------------
#pragma unroll
  for (int n = 0; n < 8; ++n) {
    int key0 = n * 16;
#pragma unroll
    for (int i = 0; i < 2; ++i) {
      int tbase = wave * 32 + 16 * i;
      int lrb = 16 * i;
      if (key0 > tbase + 15) {       // tile entirely above the diagonal
#pragma unroll
        for (int j = 0; j < 8; ++j)
          sc[(lrb + j + (half << 3)) * TDIM + key0 + col] = -1e30f;
        continue;
      }
      v8f c = vzero8();
      c = wmma_bf16(aq[i][0], load_b_bf16(Ks + key0 * HD, HD), c);
      c = wmma_bf16(aq[i][1], load_b_bf16(Ks + key0 * HD + 32, HD), c);
#pragma unroll
      for (int j = 0; j < 8; ++j) {
        int lr = lrb + j + (half << 3);
        int tg = wave * 32 + lr;
        int key = key0 + col;
        float v = c[j] * 0.125f;           // HD^-0.5
        v = 50.0f * tanhf(v * 0.02f);      // soft cap
        sc[lr * TDIM + key] = (key > tg) ? -1e30f : v;
      }
    }
  }
  __syncthreads();

  // ---- per-lane softmax of its own row; pack unnormalized exp as bf16 -----
  float inv;
  {
    float* row = sc + lane * TDIM;
    float mx = -1e30f;
    for (int j = 0; j < TDIM; ++j) mx = fmaxf(mx, row[j]);
    volatile float* vrow = row;            // in-place fp32 -> bf16 repack:
    volatile bf16* prow = (volatile bf16*)row;  // volatile defeats TBAA reorder
    float sum = 0.0f;
    for (int j = 0; j < TDIM; j += 2) {
      float e0 = __expf(vrow[j] - mx);
      float e1 = __expf(vrow[j + 1] - mx);
      sum += e0 + e1;
      prow[j] = (bf16)e0;
      prow[j + 1] = (bf16)e1;
    }
    inv = 1.0f / sum;
  }
  __syncthreads();

  // ---- O = P @ V (V staged transposed in LDS -> row-major B gather) -------
  const bf16* P = (const bf16*)sc;   // row stride = 2*TDIM bf16 elems
  v8f o[2][4];
#pragma unroll
  for (int i = 0; i < 2; ++i)
#pragma unroll
    for (int nt = 0; nt < 4; ++nt) o[i][nt] = vzero8();

#pragma unroll
  for (int i = 0; i < 2; ++i) {
    v16bf ap[4];
#pragma unroll
    for (int kk = 0; kk < 4; ++kk)
      ap[kk] = load_a_bf16(P + (size_t)(16 * i) * (2 * TDIM) + kk * 32, 2 * TDIM);
#pragma unroll
    for (int nt = 0; nt < 4; ++nt)
#pragma unroll
      for (int kk = 0; kk < 4; ++kk)
        o[i][nt] = wmma_bf16(ap[kk],
                             load_b_bf16(Vs + (nt * 16) * TDIM + kk * 32, TDIM),
                             o[i][nt]);
  }

  // ---- normalize by 1/sum (broadcast via shuffle) and store ---------------
#pragma unroll
  for (int i = 0; i < 2; ++i)
#pragma unroll
    for (int j = 0; j < 8; ++j) {
      int lr = 16 * i + j + (half << 3);
      float is = __shfl(inv, lr, 32);
      int t = wave * 32 + lr;
#pragma unroll
      for (int nt = 0; nt < 4; ++nt) {
        int c = nt * 16 + col;
        AO[((size_t)t * SDIM + s) * (NH * HD) + h * HD + c] =
            (bf16)(o[i][nt][j] * is);
      }
    }
}

// ---------------------------------------------------------------------------
// Kernel 3: output projection AO(bf16) @ WoT -> fp32 d_out, 64-row blocking.
// ---------------------------------------------------------------------------
__global__ void __launch_bounds__(32) k_out(
    const bf16* __restrict__ AO, const bf16* __restrict__ WoT,
    float* __restrict__ out) {
  int r0 = blockIdx.x << 6;
  int col0 = blockIdx.y * HD;
  v8f acc[4][4];
#pragma unroll
  for (int rt = 0; rt < 4; ++rt)
#pragma unroll
    for (int nt = 0; nt < 4; ++nt) acc[rt][nt] = vzero8();

  int lane = threadIdx.x & 31;
  for (int kk = 0; kk < KSTEPS; ++kk) {
    if (kk + 1 < KSTEPS)
      __builtin_prefetch(AO + (size_t)(r0 + lane) * NDIM + (kk + 1) * 32, 0, 3);
    v16bf a[4];
#pragma unroll
    for (int rt = 0; rt < 4; ++rt)
      a[rt] = load_a_bf16(AO + (size_t)(r0 + 16 * rt) * NDIM + kk * 32, NDIM);
#pragma unroll
    for (int nt = 0; nt < 4; ++nt) {
      v16bf b = load_b_bf16(WoT + (size_t)(col0 + nt * 16) * NDIM + kk * 32, NDIM);
#pragma unroll
      for (int rt = 0; rt < 4; ++rt)
        acc[rt][nt] = wmma_bf16(a[rt], b, acc[rt][nt]);
    }
  }
  int half = lane >> 4, col = lane & 15;
#pragma unroll
  for (int rt = 0; rt < 4; ++rt)
#pragma unroll
    for (int j = 0; j < 8; ++j) {
      int m = rt * 16 + j + (half << 3);
#pragma unroll
      for (int nt = 0; nt < 4; ++nt)
        out[(size_t)(r0 + m) * NDIM + col0 + nt * 16 + col] = acc[rt][nt][j];
    }
}

// ---------------------------------------------------------------------------
extern "C" void kernel_launch(void* const* d_in, const int* in_sizes, int n_in,
                              void* d_out, int out_size, void* d_ws, size_t ws_size,
                              hipStream_t stream) {
  (void)in_sizes; (void)n_in; (void)out_size; (void)ws_size;
  const float* x  = (const float*)d_in[0];
  const float* Wq = (const float*)d_in[1];
  const float* Wk = (const float*)d_in[2];
  const float* Wv = (const float*)d_in[3];
  const float* Wo = (const float*)d_in[4];
  const float* qw = (const float*)d_in[5];
  const float* kw = (const float*)d_in[6];
  float* out = (float*)d_out;

  char* ws = (char*)d_ws;
  size_t off = 0;
  auto carve = [&](size_t bytes) -> char* {
    char* p = ws + off;
    off += (bytes + 255) & ~(size_t)255;
    return p;
  };
  bf16* WqT = (bf16*)carve((size_t)NDIM * (NH * HD) * 2);          // 768x768
  bf16* WkT = (bf16*)carve((size_t)NDIM * (NKV * HD) * 2);         // 768x256
  bf16* WvT = (bf16*)carve((size_t)NDIM * (NKV * HD) * 2);
  bf16* WoT = (bf16*)carve((size_t)(NH * HD) * NDIM * 2);
  bf16* Qb  = (bf16*)carve((size_t)NROWS * (NH * HD) * 2);         // ~50 MB
  bf16* Kb  = (bf16*)carve((size_t)NROWS * (NKV * HD) * 2);        // ~17 MB
  bf16* Vtb = (bf16*)carve((size_t)NROWS * (NKV * HD) * 2);        // ~17 MB
  bf16* AOb = (bf16*)carve((size_t)NROWS * (NH * HD) * 2);         // ~50 MB

  int n;
  n = NDIM * NH * HD;
  k_tcvt<<<(n + 255) / 256, 256, 0, stream>>>(Wq, WqT, NDIM, NH * HD);
  n = NDIM * NKV * HD;
  k_tcvt<<<(n + 255) / 256, 256, 0, stream>>>(Wk, WkT, NDIM, NKV * HD);
  k_tcvt<<<(n + 255) / 256, 256, 0, stream>>>(Wv, WvT, NDIM, NKV * HD);
  n = NH * HD * NDIM;
  k_tcvt<<<(n + 255) / 256, 256, 0, stream>>>(Wo, WoT, NH * HD, NDIM);

  k_qkv<<<dim3(NROWS / 64, NH + 2 * NKV), 32, 0, stream>>>(
      x, WqT, WkT, WvT, qw, kw, Qb, Kb, Vtb);

  int attn_lds = 4 * 32 * TDIM * (int)sizeof(float) + 2 * TDIM * HD * 2;
  k_attn<<<dim3(SDIM, NH), 128, attn_lds, stream>>>(Qb, Kb, Vtb, AOb);

  k_out<<<dim3(NROWS / 64, NDIM / HD), 32, 0, stream>>>(AOb, WoT, out);
}
